// RoPEVisionAttention_86681029968234
// MI455X (gfx1250) — matrix operand order repaired and checked by reading, not compiled, not run on hardware
//
#include <hip/hip_runtime.h>
#include <math.h>

#define B_ 16
#define S_ 577
#define E_ 1024
#define H_ 16
#define D_ 64
#define MSIZE (B_*S_)   /* 9232 */

typedef __attribute__((ext_vector_type(16))) _Float16 v16h;
typedef __attribute__((ext_vector_type(8)))  float    v8f;
typedef __attribute__((ext_vector_type(4)))  unsigned int u32x4;
typedef __attribute__((ext_vector_type(4)))  float    f32x4;

union AF  { u32x4 u[2]; v16h v; };
union PK8 { _Float16 h[8]; u32x4 u; };

static __device__ __forceinline__ v8f wmma_f16(v16h a, v16h b, v8f c) {
  // D = A(16x32 f16) * B(32x16 f16) + C(16x16 f32)
  return __builtin_amdgcn_wmma_f32_16x16x32_f16(false, a, false, b, (short)0, c,
                                                false, false);
}

// --------------------------- TDM (Tensor Data Mover) ------------------------
#ifndef __has_builtin
#define __has_builtin(x) 0
#endif
#if __has_builtin(__builtin_amdgcn_tensor_load_to_lds) && \
    __has_builtin(__builtin_amdgcn_s_wait_tensorcnt)
#define HAVE_TDM 1
#else
#define HAVE_TDM 0
#endif
#if defined(__has_include)
#if __has_include(<hip/amd_detail/amd_gfx1250_TDM.h>)
#define TDM_6ARG 1
#else
#define TDM_6ARG 0
#endif
#else
#define TDM_6ARG 0
#endif

#if HAVE_TDM
typedef __attribute__((ext_vector_type(4))) unsigned int tdm_u4;
typedef __attribute__((ext_vector_type(4))) int tdm_i4;
typedef __attribute__((ext_vector_type(8))) int tdm_i8;

// Async DMA of a [rows=32, cols=64] f16 tile (row stride 64) from global to
// LDS, zero-filling rows beyond rows_valid (TDM OOB reads return zero).
// D# layout per CDNA5 ISA ch.8: group0 = {count/type, lds_addr, global_addr},
// group1 = {data_size, tensor dims/strides, tile dims}.
static __device__ __forceinline__ void tdm_load_k_tile(
    const void* gsrc, unsigned lds_addr, unsigned rows_valid)
{
  const unsigned long long ga = (unsigned long long)(uintptr_t)gsrc;
  tdm_u4 g0;
  g0[0] = 1u;                                       // count=1 (valid user D#)
  g0[1] = lds_addr;                                 // lds_addr (bytes)
  g0[2] = (unsigned)(ga & 0xffffffffu);             // global_addr[31:0]
  g0[3] = (unsigned)((ga >> 32) & 0x01ffffffu)      // global_addr[56:32]
        | (2u << 30);                               // type=2 ("image")
  tdm_i8 g1;
  g1[0] = (int)(1u << 16);                          // data_size=1 -> 2 bytes
  g1[1] = (int)(64u << 16);                         // tensor_dim0 = 64
  g1[2] = (int)((rows_valid & 0xffffu) << 16);      // tensor_dim1 lo16
  g1[3] = (int)((rows_valid >> 16) | (64u << 16));  // dim1 hi16 | tile_dim0=64
  g1[4] = 32;                                       // tile_dim1=32, tile_dim2=0
  g1[5] = 64;                                       // tensor_dim0_stride = 64
  g1[6] = 0;
  g1[7] = 0;
  tdm_i4 z4 = {0,0,0,0};
#if TDM_6ARG
  tdm_i8 z8 = {0,0,0,0,0,0,0,0};
  __builtin_amdgcn_tensor_load_to_lds(g0, g1, z4, z4, z8, 0);
#else
  __builtin_amdgcn_tensor_load_to_lds(g0, g1, z4, z4, 0);
#endif
}
#endif // HAVE_TDM

// ---------------------------------------------------------------------------
// Kernel 1: fused QKV projection  {q,k,v} = X*W^T + b  (+2D RoPE, q scale)
// Double-buffered LDS, one barrier per K-step, X staged once for all three.
// out: f16 [B, H, S, D] per tensor.
// ---------------------------------------------------------------------------
__global__ __launch_bounds__(256) void k_qkvf(
    const float* __restrict__ X,
    const float* __restrict__ Wq, const float* __restrict__ Wk,
    const float* __restrict__ Wv,
    const float* __restrict__ bq, const float* __restrict__ bk,
    const float* __restrict__ bv,
    _Float16* __restrict__ outq, _Float16* __restrict__ outk,
    _Float16* __restrict__ outv)
{
  __shared__ __align__(16) _Float16 Xs[2][64][32];
  __shared__ __align__(16) _Float16 Wt[2][3][64][32];
  __shared__ __align__(16) float    Cs[64][64];

  const int t = threadIdx.x;
  const int lane = t & 31, wave = t >> 5;
  const int hh = lane >> 4, nn = lane & 15;
  const int m_sub = (wave & 3) * 16;
  const int n_sub = (wave >> 2) * 32;
  const int m0 = blockIdx.x * 64;
  const int head = blockIdx.y;
  const int n0 = head * 64;
  const int srow = t >> 2;          // 0..63
  const int skk  = (t & 3) * 8;     // 0,8,16,24
  const int gms  = m0 + srow;
  const float* Wp[3] = {Wq, Wk, Wv};

  f32x4 xa, xb, wa[3], wb[3];
  auto load_step = [&](int k0) {
    f32x4 zf = {0.f, 0.f, 0.f, 0.f};
    xa = zf; xb = zf;
    if (gms < MSIZE) {
      const f32x4* p = (const f32x4*)(X + (size_t)gms * E_ + k0 + skk);
      xa = p[0]; xb = p[1];
    }
    #pragma unroll
    for (int pj = 0; pj < 3; ++pj) {
      const f32x4* pw = (const f32x4*)(Wp[pj] + (size_t)(n0 + srow) * E_ + k0 + skk);
      wa[pj] = pw[0]; wb[pj] = pw[1];
    }
  };
  auto store_step = [&](int buf) {
    PK8 pk;
    #pragma unroll
    for (int i = 0; i < 4; ++i) { pk.h[i] = (_Float16)xa[i]; pk.h[4+i] = (_Float16)xb[i]; }
    *(u32x4*)&Xs[buf][srow][skk] = pk.u;
    #pragma unroll
    for (int pj = 0; pj < 3; ++pj) {
      PK8 pw8;
      #pragma unroll
      for (int i = 0; i < 4; ++i) { pw8.h[i] = (_Float16)wa[pj][i]; pw8.h[4+i] = (_Float16)wb[pj][i]; }
      *(u32x4*)&Wt[buf][pj][srow][skk] = pw8.u;
    }
  };

  v8f cc[3][2] = {};

  load_step(0);
  store_step(0);
  __syncthreads();

  for (int kt = 0; kt < E_ / 32; ++kt) {
    const int cur = kt & 1;
    const bool more = (kt + 1) < (E_ / 32);
    if (more) load_step((kt + 1) * 32);           // overlap with WMMAs

    AF A;
    const char* xr = (const char*)&Xs[cur][m_sub + nn][0];
    A.u[0] = *(const u32x4*)(xr + 32*hh);         // K 16h..16h+7
    A.u[1] = *(const u32x4*)(xr + 32*hh + 16);    // K 16h+8..16h+15
    #pragma unroll
    for (int pj = 0; pj < 3; ++pj) {
      AF B0, B1;
      const char* w0 = (const char*)&Wt[cur][pj][n_sub + nn][0];
      B0.u[0] = *(const u32x4*)(w0 + 32*hh);      // K 16h..16h+15
      B0.u[1] = *(const u32x4*)(w0 + 32*hh + 16);
      const char* w1 = (const char*)&Wt[cur][pj][n_sub + 16 + nn][0];
      B1.u[0] = *(const u32x4*)(w1 + 32*hh);
      B1.u[1] = *(const u32x4*)(w1 + 32*hh + 16);
      cc[pj][0] = wmma_f16(A.v, B0.v, cc[pj][0]);
      cc[pj][1] = wmma_f16(A.v, B1.v, cc[pj][1]);
    }
    if (more) store_step(cur ^ 1);
    __syncthreads();
  }

  // epilogue: per projection, spill fp32 tile then bias + RoPE + f16 store
  const float*  bp[3] = {bq, bk, bv};
  _Float16*     op[3] = {outq, outk, outv};
  const int row = t >> 2;
  const int cb  = (t & 3) * 16;
  const int gm  = m0 + row;
  const int bidx = gm / S_;
  const int sidx = gm - bidx * S_;

  #pragma unroll
  for (int pj = 0; pj < 3; ++pj) {
    __syncthreads();
    #pragma unroll
    for (int j = 0; j < 8; ++j) {
      Cs[m_sub + 8*hh + j][n_sub + nn]      = cc[pj][0][j];
      Cs[m_sub + 8*hh + j][n_sub + 16 + nn] = cc[pj][1][j];
    }
    __syncthreads();
    if (gm < MSIZE) {
      const float* bias = bp[pj];
      _Float16* orow = op[pj] + (((size_t)bidx * H_ + head) * S_ + sidx) * D_;
      const int   do_rope = (pj < 2);
      const float oscale  = (pj == 0) ? 0.125f : 1.0f;  // fold 1/sqrt(64) into q
      #pragma unroll
      for (int i = 0; i < 16; ++i) {
        const int d = cb + i;
        const float x = Cs[row][d] + bias[n0 + d];
        float val = x;
        if (do_rope && sidx > 0) {
          const int p = sidx - 1;
          const float pos = (d < 32) ? (float)(p / 24) : (float)(p % 24);
          // inv_freq[i] = 10000^(-i/16); ln(10000)/16 = 0.575646273
          const float freq = __expf(-(float)(d & 15) * 0.57564627324851f);
          float sn, cs2;
          __sincosf(pos * freq, &sn, &cs2);
          const int dp = (d < 32) ? d + 32 : d - 32;
          const float xp = Cs[row][dp] + bias[n0 + dp];
          const float rot = (d < 32) ? -xp : xp;
          val = x * cs2 + rot * sn;
        }
        orow[d] = (_Float16)(val * oscale);
      }
    }
  }
}

// ---------------------------------------------------------------------------
// Kernel 2: flash attention per (b, head). Each wave: 16 query rows.
// Double-buffered 32-key K/V LDS tiles; K tile DMA'd by the TDM (wave 0),
// V tile staged transposed by the block. One barrier per key tile.
// ---------------------------------------------------------------------------
__global__ __launch_bounds__(256) void k_attn(
    const _Float16* __restrict__ Q, const _Float16* __restrict__ K,
    const _Float16* __restrict__ V, _Float16* __restrict__ O)
{
  __shared__ __align__(16) _Float16 Ks[2][32][64];     // [key][d] row-major
  __shared__ __align__(16) _Float16 Vt[2][64][32];     // [d][key] transposed
  __shared__ __align__(16) _Float16 Ps[8][16][32];     // per-wave P tile

  const int t = threadIdx.x;
  const int lane = t & 31, wave = t >> 5;
  const int hh = lane >> 4, nn = lane & 15;
  const int b = blockIdx.z;
  const int head = blockIdx.y;
  const int q0 = blockIdx.x * 128 + wave * 16;

  const size_t bh = ((size_t)b * H_ + head) * S_ * D_;
  const _Float16* qbh = Q + bh;
  const _Float16* kbh = K + bh;
  const _Float16* vbh = V + bh;

  // preload Q A-fragments (rows q0+nn, clamped; invalid rows never stored)
  int qr = q0 + nn; if (qr >= S_) qr = S_ - 1;
  const char* qrow = (const char*)(qbh + (size_t)qr * D_);
  AF qa0, qa1;
  qa0.u[0] = *(const u32x4*)(qrow + 32*hh);
  qa0.u[1] = *(const u32x4*)(qrow + 32*hh + 16);
  qa1.u[0] = *(const u32x4*)(qrow + 64 + 32*hh);
  qa1.u[1] = *(const u32x4*)(qrow + 64 + 32*hh + 16);

  v8f acc0 = {}, acc1 = {}, acc2 = {}, acc3 = {};
  float mrun[8], lrun[8];
  #pragma unroll
  for (int j = 0; j < 8; ++j) { mrun[j] = -1e30f; lrun[j] = 0.f; }

  const int skey = t >> 3;          // 0..31
  const int sdb  = (t & 7) * 8;     // 0..56

  auto stage_v = [&](int buf, int kbase) {
    const int gk = kbase + skey;
    u32x4 vd = {0, 0, 0, 0};
    if (gk < S_) {
      vd = *(const u32x4*)(vbh + (size_t)gk * D_ + sdb);
      __builtin_prefetch(vbh + (size_t)(gk + 32) * D_ + sdb, 0, 3);
    }
    PK8 vv; vv.u = vd;
    #pragma unroll
    for (int i = 0; i < 8; ++i) Vt[buf][sdb + i][skey] = vv.h[i];
  };
  auto stage_k = [&](int buf, int kbase) {
#if HAVE_TDM
    if (wave == 0)
      tdm_load_k_tile(kbh + (size_t)kbase * D_,
                      (unsigned)(uintptr_t)&Ks[buf][0][0],
                      (unsigned)(S_ - kbase));
#else
    const int gk = kbase + skey;
    u32x4 kd = {0, 0, 0, 0};
    if (gk < S_) kd = *(const u32x4*)(kbh + (size_t)gk * D_ + sdb);
    *(u32x4*)&Ks[buf][skey][sdb] = kd;
#endif
  };

  const int NTILES = (S_ + 31) / 32;      // 19
  stage_v(0, 0);
  stage_k(0, 0);
#if HAVE_TDM
  if (wave == 0) __builtin_amdgcn_s_wait_tensorcnt(0);
#endif
  __syncthreads();

  for (int kt = 0; kt < NTILES; ++kt) {
    const int cur = kt & 1;
    const int kbase = kt * 32;
    const bool more = (kt + 1) < NTILES;
    if (more) { stage_v(cur ^ 1, kbase + 32); stage_k(cur ^ 1, kbase + 32); }

    // scores S = Q * K^T : two 16x16 col-fragments, K-dim 64 = 2 chunks
    v8f s0 = {}, s1 = {};
    {
      AF kb;
      const char* kr0 = (const char*)&Ks[cur][nn][0];
      kb.u[0] = *(const u32x4*)(kr0 + 32*hh);
      kb.u[1] = *(const u32x4*)(kr0 + 32*hh + 16);
      s0 = wmma_f16(qa0.v, kb.v, s0);
      kb.u[0] = *(const u32x4*)(kr0 + 64 + 32*hh);
      kb.u[1] = *(const u32x4*)(kr0 + 64 + 32*hh + 16);
      s0 = wmma_f16(qa1.v, kb.v, s0);
      const char* kr1 = (const char*)&Ks[cur][16 + nn][0];
      kb.u[0] = *(const u32x4*)(kr1 + 32*hh);
      kb.u[1] = *(const u32x4*)(kr1 + 32*hh + 16);
      s1 = wmma_f16(qa0.v, kb.v, s1);
      kb.u[0] = *(const u32x4*)(kr1 + 64 + 32*hh);
      kb.u[1] = *(const u32x4*)(kr1 + 64 + 32*hh + 16);
      s1 = wmma_f16(qa1.v, kb.v, s1);
    }

    // mask + online softmax (row r = j + 8*hh; 16-lane butterflies)
    const bool inval0 = (kbase + nn)      >= S_;
    const bool inval1 = (kbase + 16 + nn) >= S_;
    #pragma unroll
    for (int j = 0; j < 8; ++j) {
      float a0 = inval0 ? -1e30f : s0[j];
      float a1 = inval1 ? -1e30f : s1[j];
      float mt = fmaxf(a0, a1);
      mt = fmaxf(mt, __shfl_xor(mt, 1, 32));
      mt = fmaxf(mt, __shfl_xor(mt, 2, 32));
      mt = fmaxf(mt, __shfl_xor(mt, 4, 32));
      mt = fmaxf(mt, __shfl_xor(mt, 8, 32));
      const float mnew  = fmaxf(mrun[j], mt);
      const float alpha = __expf(mrun[j] - mnew);
      mrun[j] = mnew;
      const float p0 = __expf(a0 - mnew);
      const float p1 = __expf(a1 - mnew);
      float rs = p0 + p1;
      rs += __shfl_xor(rs, 1, 32);
      rs += __shfl_xor(rs, 2, 32);
      rs += __shfl_xor(rs, 4, 32);
      rs += __shfl_xor(rs, 8, 32);
      lrun[j] = lrun[j] * alpha + rs;
      acc0[j] *= alpha; acc1[j] *= alpha; acc2[j] *= alpha; acc3[j] *= alpha;
      Ps[wave][8*hh + j][nn]      = (_Float16)p0;   // C-layout -> LDS
      Ps[wave][8*hh + j][16 + nn] = (_Float16)p1;
    }
    // per-wave producer->consumer on Ps: hardware keeps same-wave LDS ops in
    // order; this compiler fence stops reordering of the type-punned reload.
    asm volatile("" ::: "memory");

    // O += P(16x32) * V(32x64): one A-frag, four B-frags from Vt
    AF pa;
    const char* pr = (const char*)&Ps[wave][nn][0];
    pa.u[0] = *(const u32x4*)(pr + 32*hh);
    pa.u[1] = *(const u32x4*)(pr + 32*hh + 16);
    AF vb;
    const char* vr0 = (const char*)&Vt[cur][nn][0];
    vb.u[0] = *(const u32x4*)(vr0 + 32*hh);
    vb.u[1] = *(const u32x4*)(vr0 + 32*hh + 16);
    acc0 = wmma_f16(pa.v, vb.v, acc0);
    const char* vr1 = (const char*)&Vt[cur][16 + nn][0];
    vb.u[0] = *(const u32x4*)(vr1 + 32*hh);
    vb.u[1] = *(const u32x4*)(vr1 + 32*hh + 16);
    acc1 = wmma_f16(pa.v, vb.v, acc1);
    const char* vr2 = (const char*)&Vt[cur][32 + nn][0];
    vb.u[0] = *(const u32x4*)(vr2 + 32*hh);
    vb.u[1] = *(const u32x4*)(vr2 + 32*hh + 16);
    acc2 = wmma_f16(pa.v, vb.v, acc2);
    const char* vr3 = (const char*)&Vt[cur][48 + nn][0];
    vb.u[0] = *(const u32x4*)(vr3 + 32*hh);
    vb.u[1] = *(const u32x4*)(vr3 + 32*hh + 16);
    acc3 = wmma_f16(pa.v, vb.v, acc3);

#if HAVE_TDM
    if (more && wave == 0) __builtin_amdgcn_s_wait_tensorcnt(0);
#endif
    __syncthreads();
  }

  // normalize and store: o[b][s][head*64 + d]
  #pragma unroll
  for (int j = 0; j < 8; ++j) {
    const int r = q0 + 8*hh + j;
    if (r < S_) {
      _Float16* orow = O + ((size_t)b * S_ + r) * E_ + head * D_;
      const float inv = 1.0f / lrun[j];
      orow[nn]      = (_Float16)(acc0[j] * inv);
      orow[16 + nn] = (_Float16)(acc1[j] * inv);
      orow[32 + nn] = (_Float16)(acc2[j] * inv);
      orow[48 + nn] = (_Float16)(acc3[j] * inv);
    }
  }
}

// ---------------------------------------------------------------------------
// Kernel 3: output projection  out = o * Wo^T + bo  (f16 acts, f32 out)
// Double-buffered LDS, one barrier per K-step.
// ---------------------------------------------------------------------------
__global__ __launch_bounds__(256) void k_oproj(
    const _Float16* __restrict__ Xh, const float* __restrict__ W,
    const float* __restrict__ bias, float* __restrict__ out)
{
  __shared__ __align__(16) _Float16 Xs[2][64][32];
  __shared__ __align__(16) _Float16 Wt[2][64][32];

  const int t = threadIdx.x;
  const int lane = t & 31, wave = t >> 5;
  const int hh = lane >> 4, nn = lane & 15;
  const int m_sub = (wave & 3) * 16;
  const int n_sub = (wave >> 2) * 32;
  const int m0 = blockIdx.x * 64;
  const int n0 = blockIdx.y * 64;
  const int srow = t >> 2;
  const int skk  = (t & 3) * 8;
  const int gms  = m0 + srow;

  u32x4 xd; f32x4 wa, wb;
  auto load_step = [&](int k0) {
    u32x4 z = {0, 0, 0, 0};
    xd = z;
    if (gms < MSIZE) xd = *(const u32x4*)(Xh + (size_t)gms * E_ + k0 + skk);
    const f32x4* pw = (const f32x4*)(W + (size_t)(n0 + srow) * E_ + k0 + skk);
    wa = pw[0]; wb = pw[1];
  };
  auto store_step = [&](int buf) {
    *(u32x4*)&Xs[buf][srow][skk] = xd;
    PK8 pw8;
    #pragma unroll
    for (int i = 0; i < 4; ++i) { pw8.h[i] = (_Float16)wa[i]; pw8.h[4+i] = (_Float16)wb[i]; }
    *(u32x4*)&Wt[buf][srow][skk] = pw8.u;
  };

  v8f c0 = {}, c1 = {};
  load_step(0);
  store_step(0);
  __syncthreads();

  for (int kt = 0; kt < E_ / 32; ++kt) {
    const int cur = kt & 1;
    const bool more = (kt + 1) < (E_ / 32);
    if (more) load_step((kt + 1) * 32);

    AF A, B0, B1;
    const char* xr = (const char*)&Xs[cur][m_sub + nn][0];
    A.u[0] = *(const u32x4*)(xr + 32*hh);
    A.u[1] = *(const u32x4*)(xr + 32*hh + 16);
    const char* w0 = (const char*)&Wt[cur][n_sub + nn][0];
    B0.u[0] = *(const u32x4*)(w0 + 32*hh);
    B0.u[1] = *(const u32x4*)(w0 + 32*hh + 16);
    const char* w1 = (const char*)&Wt[cur][n_sub + 16 + nn][0];
    B1.u[0] = *(const u32x4*)(w1 + 32*hh);
    B1.u[1] = *(const u32x4*)(w1 + 32*hh + 16);
    c0 = wmma_f16(A.v, B0.v, c0);
    c1 = wmma_f16(A.v, B1.v, c1);

    if (more) store_step(cur ^ 1);
    __syncthreads();
  }

  #pragma unroll
  for (int j = 0; j < 8; ++j) {
    const int gm = m0 + m_sub + 8*hh + j;
    if (gm < MSIZE) {
      const int col0 = n0 + n_sub + nn;
      out[(size_t)gm * E_ + col0]      = c0[j] + bias[col0];
      out[(size_t)gm * E_ + col0 + 16] = c1[j] + bias[col0 + 16];
    }
  }
}

// ---------------------------------------------------------------------------
extern "C" void kernel_launch(void* const* d_in, const int* in_sizes, int n_in,
                              void* d_out, int out_size, void* d_ws, size_t ws_size,
                              hipStream_t stream) {
  const float* hs = (const float*)d_in[0];
  const float* Wq = (const float*)d_in[1];
  const float* bq = (const float*)d_in[2];
  const float* Wk = (const float*)d_in[3];
  const float* bk = (const float*)d_in[4];
  const float* Wv = (const float*)d_in[5];
  const float* bv = (const float*)d_in[6];
  const float* Wo = (const float*)d_in[7];
  const float* bo = (const float*)d_in[8];
  float* out = (float*)d_out;

  const size_t qkvElems = (size_t)B_ * H_ * S_ * D_;   // 9,453,568 per tensor
  _Float16* qW = (_Float16*)d_ws;
  _Float16* kW = qW + qkvElems;
  _Float16* vW = kW + qkvElems;
  _Float16* oW = vW + qkvElems;                        // B*S*E elements

  dim3 blk(256);
  dim3 g1((MSIZE + 63) / 64, H_);                      // 145 x 16
  k_qkvf<<<g1, blk, 0, stream>>>(hs, Wq, Wk, Wv, bq, bk, bv, qW, kW, vW);

  dim3 g2((S_ + 127) / 128, H_, B_);                   // 5 x 16 x 16
  k_attn<<<g2, blk, 0, stream>>>(qW, kW, vW, oW);

  k_oproj<<<g1, blk, 0, stream>>>(oW, Wo, bo, out);
}
